// RSSM_51135880626707
// MI455X (gfx1250) — matrix operand order repaired and checked by reading, not compiled, not run on hardware
//
#include <hip/hip_runtime.h>

typedef unsigned short u16;
typedef unsigned int   u32;

typedef __bf16 bf16v16 __attribute__((ext_vector_type(16)));
typedef float  f32v8   __attribute__((ext_vector_type(8)));
typedef u32    u32x4   __attribute__((ext_vector_type(4)));
typedef int    i32x4v  __attribute__((ext_vector_type(4)));

// gfx1250 async global->LDS copy engine (ASYNCcnt-tracked), if the toolchain
// declares the builtin; otherwise fall back to vector copy through VGPRs.
#if defined(__has_builtin)
#  if __has_builtin(__builtin_amdgcn_global_load_async_to_lds_b128)
#    define HAVE_ASYNC_LDS 1
#  else
#    define HAVE_ASYNC_LDS 0
#  endif
#  if __has_builtin(__builtin_amdgcn_s_wait_asynccnt)
#    define HAVE_WAIT_ASYNC 1
#  else
#    define HAVE_WAIT_ASYNC 0
#  endif
#else
#  define HAVE_ASYNC_LDS 0
#  define HAVE_WAIT_ASYNC 0
#endif

#if HAVE_ASYNC_LDS
typedef __attribute__((address_space(1))) i32x4v* as1_i32x4p;   // global (prints as __device__)
typedef __attribute__((address_space(3))) i32x4v* as3_i32x4p;   // LDS
#endif

namespace {
constexpr int kB = 32, kT = 64, kEMBED = 256;
constexpr int kDETER = 512, kHID = 512;
constexpr int kD = 64, kC = 64, kST = kD * kC;          // 4096
constexpr int kOutW = kDETER + 4 * kST;                  // 16896
constexpr int kBlocks = 8, kThreads = 256;               // 64 wave32s total
constexpr int kBarSlots = 512;                           // 6 barriers/step * 64 steps = 384 used
}

// ---------- helpers ----------
__device__ __forceinline__ u16 f2bf(float f) {
  u32 u = __builtin_bit_cast(u32, f);
  u32 r = (u + 0x7FFFu + ((u >> 16) & 1u)) >> 16;        // round-to-nearest-even
  return (u16)r;
}
__device__ __forceinline__ float sigmoidf_(float x) { return 1.f / (1.f + __expf(-x)); }

__device__ __forceinline__ float gumbelf(u32 t, u32 w, u32 b, u32 d, u32 c) {
  u32 x = t * 0x9E3779B1u ^ (w * 0x85EBCA77u + b * 0xC2B2AE3Du + d * 0x27D4EB2Fu + c * 0x165667B1u) ^ 0xA5A5A5A5u;
  x ^= x >> 16; x *= 0x7FEB352Du; x ^= x >> 15; x *= 0x846CA68Bu; x ^= x >> 16;
  float u = ((x >> 8) + 0.5f) * (1.0f / 16777216.0f);    // (0,1)
  return -__logf(-__logf(u));
}

// Single-use global barrier slots: each slot goes 0 -> kBlocks exactly once per call.
// Prologue kernel re-zeroes slots every call, so graph replay is deterministic.
__device__ __forceinline__ void gbar(u32* slots, int idx, int nblocks) {
  __threadfence();
  __syncthreads();
  if (threadIdx.x == 0) {
    u32* p = slots + idx;
    __hip_atomic_fetch_add(p, 1u, __ATOMIC_RELEASE, __HIP_MEMORY_SCOPE_AGENT);
    while (__hip_atomic_load(p, __ATOMIC_ACQUIRE, __HIP_MEMORY_SCOPE_AGENT) < (u32)nblocks) {
      __builtin_amdgcn_s_sleep(1);
    }
  }
  __syncthreads();
  __threadfence();
}

// Cooperative copy of an activation panel (global -> LDS), 16B per lane per issue.
__device__ __forceinline__ void stage_lds(u16* __restrict__ dst, const u16* __restrict__ src,
                                          int nElems) {
#if HAVE_ASYNC_LDS
  for (int i = (int)threadIdx.x * 8; i < nElems; i += kThreads * 8) {
    __builtin_amdgcn_global_load_async_to_lds_b128(
        (as1_i32x4p)(src + i), (as3_i32x4p)(dst + i), 0, 0);
  }
#else
  for (int i = (int)threadIdx.x * 8; i < nElems; i += kThreads * 8) {
    *(u32x4*)(dst + i) = *(const u32x4*)(src + i);
  }
#endif
}

// Wait for this wave's outstanding async LDS fills before the workgroup barrier.
__device__ __forceinline__ void stage_fence() {
#if HAVE_ASYNC_LDS
#  if HAVE_WAIT_ASYNC
  __builtin_amdgcn_s_wait_asynccnt(0);
#  else
  asm volatile("s_wait_asynccnt 0" ::: "memory");
#  endif
#endif
  __syncthreads();
}

// ---------- bf16 WMMA GEMM: Out[32,N] = A[32,K](bf16) x Bpacked[K,N](bf16), f32 acc ----------
// A: row-major bf16 (LDS-staged or global), row stride Kpad, 16B-aligned rows.
// Bp: pre-packed fragment-major: chunk (kc, ntile) = 32 lanes x 16 bf16 (two b128 loads/lane).
// Both m-tiles (M=32 -> 2) are processed per n-tile so every B fragment is loaded once
// and feeds two v_wmma (B traffic halved; B has no reuse across waves, so LDS/TDM
// staging of B would buy nothing - direct pipelined global->VGPR is the right path).
template <typename Epi>
__device__ __forceinline__ void gemm_bf16(const u16* __restrict__ A, int Kpad,
                                          const u16* __restrict__ Bp, int K, int N,
                                          int waveId, int nWaves, Epi&& epi) {
  const int NT = N >> 4;
  const int KC = K >> 5;
  const int lane = (int)(threadIdx.x & 31u);
  const int lan15 = lane & 15;
  const int hi = (lane >> 4) & 1;             // which K-half this lane holds (A layout)
  const size_t bstride = (size_t)NT * 512;    // packed elems per kc chunk
  for (int nt = waveId; nt < NT; nt += nWaves) {
    const u16* arow0 = A + (size_t)lan15 * Kpad + (hi << 3);
    const u16* arow1 = arow0 + (size_t)16 * Kpad;
    const u16* brow = Bp + (((size_t)nt * 32 + lane) << 4);
    f32v8 acc0 = {}, acc1 = {};
    for (int kc = 0; kc < KC; ++kc) {
      union { bf16v16 v; u32x4 q[2]; } a0, a1, bf;
      const u32x4* ap0 = (const u32x4*)(arow0 + (kc << 5));
      const u32x4* ap1 = (const u32x4*)(arow1 + (kc << 5));
      a0.q[0] = ap0[0]; a0.q[1] = ap0[2];     // K = base+0..7, base+16..23
      a1.q[0] = ap1[0]; a1.q[1] = ap1[2];
      const u32x4* bp = (const u32x4*)(brow + kc * bstride);
      bf.q[0] = bp[0]; bf.q[1] = bp[1];
      acc0 = __builtin_amdgcn_wmma_f32_16x16x32_bf16(false, a0.v, false, bf.v,
                                                     (short)0, acc0, false, false);
      acc1 = __builtin_amdgcn_wmma_f32_16x16x32_bf16(false, a1.v, false, bf.v,
                                                     (short)0, acc1, false, false);
    }
    const int mb = hi << 3;
    const int n = (nt << 4) + lan15;
#pragma unroll
    for (int j = 0; j < 8; ++j) {
      epi(mb + j, n, acc0[j]);
      epi(16 + mb + j, n, acc1[j]);
    }
  }
}

// ---------- prologue: pack weights, transpose W_pre, convert obs, zero state ----------
__device__ void pack_w(const float* __restrict__ W, u16* __restrict__ dst,
                       int K, int N, int ldw, int gtid, int nth) {
  const int NT = N >> 4;
  const int total = K * N;
  for (int g = gtid; g < total; g += nth) {
    const int e = g & 15;
    const int lane = (g >> 4) & 31;
    const int rest = g >> 9;
    const int nt = rest % NT;
    const int kc = rest / NT;
    const int k = (kc << 5) + ((lane & 16) ? 8 : 0) + (e & 7) + ((e & 8) ? 16 : 0);   // B layout: table order, K = 8h+e / 16+8h+(e-8)
    const int n = (nt << 4) + (lane & 15);
    dst[g] = f2bf(W[(size_t)n * ldw + k]);                  // B[k][n] = W[n][k]
  }
}

struct InitArgs {
  const float *obs, *Wpre, *Wih, *Whh, *Wp1, *Wp2, *Wq1, *Wq2;
  u32* bars;
  float *hf, *WpreT;
  u16 *hbf, *obsbf;
  u16 *ihP, *hhP, *p1P, *p2P, *q1P, *q2P;
};

__global__ void rssm_init(InitArgs a) {
  const int gtid = (int)(blockIdx.x * blockDim.x + threadIdx.x);
  const int nth = (int)(gridDim.x * blockDim.x);
  for (int g = gtid; g < kBarSlots; g += nth) a.bars[g] = 0u;
  for (int g = gtid; g < kB * kDETER; g += nth) { a.hf[g] = 0.f; a.hbf[g] = 0; }
  for (int g = gtid; g < kB * kT * kEMBED; g += nth) a.obsbf[g] = f2bf(a.obs[g]);
  // W_pre transposed to f32 [4096][512]: S1 is a gather-sum of one-hot-selected rows
  // (exact: summing the zero terms of the reference GEMM contributes nothing).
  for (int g = gtid; g < kST * kHID; g += nth) {
    const int k = g >> 9, n = g & 511;
    a.WpreT[g] = a.Wpre[(size_t)n * (kST + 2) + k];
  }
  pack_w(a.Wih, a.ihP, kHID, 3 * kDETER, kHID, gtid, nth);
  pack_w(a.Whh, a.hhP, kDETER, 3 * kDETER, kDETER, gtid, nth);
  pack_w(a.Wp1, a.p1P, kDETER, kHID, kDETER, gtid, nth);
  pack_w(a.Wp2, a.p2P, kHID, kST, kHID, gtid, nth);
  pack_w(a.Wq1, a.q1P, kDETER + kEMBED, kHID, kDETER + kEMBED, gtid, nth);
  pack_w(a.Wq2, a.q2P, kHID, kST, kHID, gtid, nth);
}

// ---------- persistent cooperative RSSM scan ----------
struct MainArgs {
  const float *act, *Wpre, *WpreT;
  const float *b_pre, *b_ih, *b_hh, *b_p1, *b_p2, *b_q1, *b_q2;
  float* out;
  u32* bars;
  float *hf, *gx, *gh;
  u16 *xbf, *hbf, *q1A, *p1a, *q1a, *zidx;
  const u16 *obsbf, *ihP, *hhP, *p1P, *p2P, *q1P, *q2P;
  const void* pfBase; size_t pfBytes;
};

__launch_bounds__(kThreads, 1)
__global__ void rssm_main(MainArgs p) {
  __shared__ __align__(16) u16 sA[32768];    // 64 KB A-panel staging (of 320 KB/WGP)
  const int tid = (int)threadIdx.x;
  const int gtid = (int)(blockIdx.x * blockDim.x) + tid;
  const int nth = kBlocks * kThreads;
  const int waveId = gtid >> 5;
  const int nWaves = nth >> 5;
  int bar = 0;

  // Warm L2 with the full packed-weight region before the serial scan starts
  // (global_prefetch_b8; speculative, no counters).
  {
    const char* base = (const char*)p.pfBase;
    for (size_t o = (size_t)gtid * 128; o < p.pfBytes; o += (size_t)nth * 128)
      __builtin_prefetch(base + o, 0, 1);
  }

  for (int t = 0; t < kT; ++t) {
    // S1: x = silu([zst, a] @ Wpre.T + b_pre). zst is one-hot -> gather-sum of the 64
    // selected rows of WpreT (coalesced over n), + rank-2 f32 action correction.
    for (int i = gtid; i < kB * kHID; i += nth) {
      const int b = i >> 9, n = i & 511;
      const float a0 = p.act[(b * kT + t) * 2 + 0];
      const float a1 = p.act[(b * kT + t) * 2 + 1];
      float v = p.b_pre[n]
              + a0 * p.Wpre[(size_t)n * (kST + 2) + kST]
              + a1 * p.Wpre[(size_t)n * (kST + 2) + kST + 1];
      if (t > 0) {
        const u16* zi = p.zidx + b * kD;
#pragma unroll 4
        for (int d = 0; d < kD; ++d)
          v += p.WpreT[(size_t)zi[d] * kHID + n];
      }
      v = v * sigmoidf_(v);
      p.xbf[i] = f2bf(v);
    }
    gbar(p.bars, bar++, kBlocks);

    // S2: gx = x @ Wih.T + b_ih ; gh = h @ Whh.T + b_hh   (A panels from LDS)
    stage_lds(sA, p.xbf, kB * kHID);
    stage_lds(sA + kB * kHID, p.hbf, kB * kDETER);
    stage_fence();
    gemm_bf16(sA, kHID, p.ihP, kHID, 3 * kDETER, waveId, nWaves,
      [&](int m, int n, float acc) { p.gx[m * 3 * kDETER + n] = acc + p.b_ih[n]; });
    gemm_bf16(sA + kB * kHID, kDETER, p.hhP, kDETER, 3 * kDETER, waveId, nWaves,
      [&](int m, int n, float acc) { p.gh[m * 3 * kDETER + n] = acc + p.b_hh[n]; });
    gbar(p.bars, bar++, kBlocks);

    // S3: GRU elementwise + build q1 input [h2 | o]
    for (int i = gtid; i < kB * kDETER; i += nth) {
      const int b = i >> 9, j = i & 511;
      const float* gxr = p.gx + b * 3 * kDETER;
      const float* ghr = p.gh + b * 3 * kDETER;
      const float r  = sigmoidf_(gxr[j] + ghr[j]);
      const float uu = sigmoidf_(gxr[kDETER + j] + ghr[kDETER + j]);
      const float nn = tanhf(gxr[2 * kDETER + j] + r * ghr[2 * kDETER + j]);
      const float h2 = (1.f - uu) * nn + uu * p.hf[i];
      p.hf[i] = h2;
      const u16 hb = f2bf(h2);
      p.hbf[i] = hb;
      p.q1A[b * (kDETER + kEMBED) + j] = hb;
      p.out[(size_t)(b * kT + t) * kOutW + j] = h2;      // deter
    }
    for (int i = gtid; i < kB * kEMBED; i += nth) {
      const int b = i >> 8, e = i & 255;
      p.q1A[b * (kDETER + kEMBED) + kDETER + e] = p.obsbf[(size_t)(b * kT + t) * kEMBED + e];
    }
    gbar(p.bars, bar++, kBlocks);

    // S4: p1a = silu(h2 @ Wp1.T + b) ; q1a = silu([h2,o] @ Wq1.T + b)
    stage_lds(sA, p.hbf, kB * kDETER);                   // fresh h2 panel
    stage_fence();
    gemm_bf16(sA, kDETER, p.p1P, kDETER, kHID, waveId, nWaves,
      [&](int m, int n, float acc) {
        float v = acc + p.b_p1[n]; v = v * sigmoidf_(v);
        p.p1a[m * kHID + n] = f2bf(v);
      });
    gemm_bf16(p.q1A, kDETER + kEMBED, p.q1P, kDETER + kEMBED, kHID, waveId, nWaves,
      [&](int m, int n, float acc) {
        float v = acc + p.b_q1[n]; v = v * sigmoidf_(v);
        p.q1a[m * kHID + n] = f2bf(v);
      });
    gbar(p.bars, bar++, kBlocks);

    // S5: logits pl = p1a @ Wp2.T + b ; ql = q1a @ Wq2.T + b  (written straight to out)
    stage_lds(sA, p.p1a, kB * kHID);
    stage_lds(sA + kB * kHID, p.q1a, kB * kHID);
    stage_fence();
    gemm_bf16(sA, kHID, p.p2P, kHID, kST, waveId, nWaves,
      [&](int m, int n, float acc) {
        p.out[(size_t)(m * kT + t) * kOutW + (kDETER + 2 * kST) + n] = acc + p.b_p2[n];
      });
    gemm_bf16(sA + kB * kHID, kHID, p.q2P, kHID, kST, waveId, nWaves,
      [&](int m, int n, float acc) {
        p.out[(size_t)(m * kT + t) * kOutW + (kDETER + 3 * kST) + n] = acc + p.b_q2[n];
      });
    gbar(p.bars, bar++, kBlocks);

    // S6: straight-through categorical sample (forward value == one-hot);
    // posterior also records selected column indices for next step's S1 gather.
    for (int g = gtid; g < 2 * kB * kD; g += nth) {
      const int which = (g >= kB * kD) ? 1 : 0;            // 0=prior, 1=posterior
      const int rr = g & (kB * kD - 1);
      const int b = rr >> 6, d = rr & 63;
      const size_t row = (size_t)(b * kT + t) * kOutW;
      const float* L = p.out + row + (which ? (kDETER + 3 * kST) : (kDETER + 2 * kST)) + d * kC;
      float l[kC];
#pragma unroll
      for (int c = 0; c < kC; ++c) l[c] = L[c];            // one load pass, 3 reg passes
      float m1 = -3.4e38f;
#pragma unroll
      for (int c = 0; c < kC; ++c) m1 = fmaxf(m1, l[c]);
      float s = 0.f;
#pragma unroll
      for (int c = 0; c < kC; ++c) s += __expf(l[c] - m1);
      const float inv = 1.f / s;
      float best = -3.4e38f; int bc = 0;
#pragma unroll
      for (int c = 0; c < kC; ++c) {
        const float pp = 0.99f * __expf(l[c] - m1) * inv + (0.01f / kC);
        const float sc = __logf(pp) + gumbelf((u32)t, (u32)which, (u32)b, (u32)d, (u32)c);
        if (sc > best) { best = sc; bc = c; }
      }
      float* O = p.out + row + (which ? (kDETER + kST) : kDETER) + d * kC;
#pragma unroll
      for (int c = 0; c < kC; ++c) O[c] = (c == bc) ? 1.f : 0.f;
      if (which) p.zidx[b * kD + d] = (u16)(d * kC + bc);
    }
    gbar(p.bars, bar++, kBlocks);
  }
}

// ---------- host ----------
extern "C" void kernel_launch(void* const* d_in, const int* in_sizes, int n_in,
                              void* d_out, int out_size, void* d_ws, size_t ws_size,
                              hipStream_t stream) {
  const float* obs  = (const float*)d_in[0];
  const float* act  = (const float*)d_in[1];
  const float* Wpre = (const float*)d_in[2];
  const float* bpre = (const float*)d_in[3];
  const float* Wih  = (const float*)d_in[4];
  const float* bih  = (const float*)d_in[5];
  const float* Whh  = (const float*)d_in[6];
  const float* bhh  = (const float*)d_in[7];
  const float* Wp1  = (const float*)d_in[8];
  const float* bp1  = (const float*)d_in[9];
  const float* Wp2  = (const float*)d_in[10];
  const float* bp2  = (const float*)d_in[11];
  const float* Wq1  = (const float*)d_in[12];
  const float* bq1  = (const float*)d_in[13];
  const float* Wq2  = (const float*)d_in[14];
  const float* bq2  = (const float*)d_in[15];

  char* w = (char*)d_ws;
  size_t off = 0;
  auto carve = [&](size_t bytes) {
    void* pt = w + off;
    off = (off + bytes + 255) & ~(size_t)255;
    return pt;
  };
  u32*   bars  = (u32*)  carve((size_t)kBarSlots * 4);
  float* hf    = (float*)carve((size_t)kB * kDETER * 4);
  float* gx    = (float*)carve((size_t)kB * 3 * kDETER * 4);
  float* gh    = (float*)carve((size_t)kB * 3 * kDETER * 4);
  u16*   xbf   = (u16*)  carve((size_t)kB * kHID * 2);
  u16*   hbf   = (u16*)  carve((size_t)kB * kDETER * 2);
  u16*   q1A   = (u16*)  carve((size_t)kB * (kDETER + kEMBED) * 2);
  u16*   p1a   = (u16*)  carve((size_t)kB * kHID * 2);
  u16*   q1a   = (u16*)  carve((size_t)kB * kHID * 2);
  u16*   zidx  = (u16*)  carve((size_t)kB * kD * 2);
  u16*   obsbf = (u16*)  carve((size_t)kB * kT * kEMBED * 2);
  const size_t offW = off;                                  // prefetch region start
  float* WpreT = (float*)carve((size_t)kST * kHID * 4);
  u16*   ihP   = (u16*)  carve((size_t)kHID * 3 * kDETER * 2);
  u16*   hhP   = (u16*)  carve((size_t)kDETER * 3 * kDETER * 2);
  u16*   p1P   = (u16*)  carve((size_t)kDETER * kHID * 2);
  u16*   p2P   = (u16*)  carve((size_t)kHID * kST * 2);
  u16*   q1P   = (u16*)  carve((size_t)(kDETER + kEMBED) * kHID * 2);
  u16*   q2P   = (u16*)  carve((size_t)kHID * kST * 2);
  const size_t pfBytes = off - offW;                        // ~21 MB weight region

  InitArgs ia{obs, Wpre, Wih, Whh, Wp1, Wp2, Wq1, Wq2,
              bars, hf, WpreT, hbf, obsbf,
              ihP, hhP, p1P, p2P, q1P, q2P};
  rssm_init<<<dim3(1024), dim3(256), 0, stream>>>(ia);

  MainArgs ma{act, Wpre, WpreT, bpre, bih, bhh, bp1, bp2, bq1, bq2,
              (float*)d_out, bars, hf, gx, gh,
              xbf, hbf, q1A, p1a, q1a, zidx,
              obsbf, ihP, hhP, p1P, p2P, q1P, q2P,
              (const void*)WpreT, pfBytes};
  rssm_main<<<dim3(kBlocks), dim3(kThreads), 0, stream>>>(ma);

  (void)in_sizes; (void)n_in; (void)out_size; (void)ws_size;
}
